// LigerLMHeadCPO_48859547959894
// MI455X (gfx1250) — compile-verified, matrix operand order checked by reading, not run on hardware
//
#include <hip/hip_runtime.h>
#include <hip/hip_bf16.h>
#include <math.h>

#define H        4096
#define V        32000
#define T_TOTAL  4096
#define TM       128
#define TN       128
#define KSTEP    32
#define NSTEPS   (H / KSTEP)    // 128
#define AS_STRIDE 40            // ushorts per LDS row: 32 data + 8 pad (80B) -> conflict-free b128 frag reads
#define NGROUPS  16
#define NTILES   250            // 32000 / 128
#define TILES_PER_GROUP 16

typedef __attribute__((ext_vector_type(16))) __bf16 v16bf;
typedef __attribute__((ext_vector_type(8)))  __bf16 v8bf;
typedef __attribute__((ext_vector_type(8)))  float  v8f;

__device__ __forceinline__ unsigned short f2bf(float f) {
  unsigned u = __float_as_uint(f);
  unsigned r = u + 0x7FFFu + ((u >> 16) & 1u);   // round-to-nearest-even
  return (unsigned short)(r >> 16);
}

// Load one 16x32 bf16 WMMA fragment row-slice for this lane from LDS.
__device__ __forceinline__ v16bf ldsFrag(const unsigned short* base, int row, int half) {
  const v8bf lo = *reinterpret_cast<const v8bf*>(base + row * AS_STRIDE + half * 8);
  const v8bf hi = *reinterpret_cast<const v8bf*>(base + row * AS_STRIDE + 16 + half * 8);
  return __builtin_shufflevector(lo, hi, 0,1,2,3,4,5,6,7,8,9,10,11,12,13,14,15);
}

// Fetch one K-step worth of A/B f32 data into registers (4 float4 each).
__device__ __forceinline__ void global_fetch(
    const float* __restrict__ x, const float* __restrict__ W,
    int tok0, int col0, int k0, int tid, float4 a[4], float4 b[4]) {
  #pragma unroll
  for (int i = 0; i < 4; ++i) {
    const int c  = i * 256 + tid;
    const int r  = c >> 3;
    const int f4 = c & 7;
    a[i] = *reinterpret_cast<const float4*>(x + (size_t)(tok0 + r) * H + k0 + f4 * 4);
    b[i] = *reinterpret_cast<const float4*>(W + (size_t)(col0 + r) * H + k0 + f4 * 4);
  }
}

// Convert prefetched registers to bf16 and store into an LDS buffer pair.
__device__ __forceinline__ void lds_store(
    unsigned short* As, unsigned short* Bs, int tid,
    const float4 a[4], const float4 b[4]) {
  #pragma unroll
  for (int i = 0; i < 4; ++i) {
    const int c  = i * 256 + tid;
    const int r  = c >> 3;
    const int f4 = c & 7;
    ushort4 ap; ap.x = f2bf(a[i].x); ap.y = f2bf(a[i].y); ap.z = f2bf(a[i].z); ap.w = f2bf(a[i].w);
    *reinterpret_cast<ushort4*>(&As[r * AS_STRIDE + f4 * 4]) = ap;
    ushort4 bp; bp.x = f2bf(b[i].x); bp.y = f2bf(b[i].y); bp.z = f2bf(b[i].z); bp.w = f2bf(b[i].w);
    *reinterpret_cast<ushort4*>(&Bs[r * AS_STRIDE + f4 * 4]) = bp;
  }
}

// ---------------------------------------------------------------------------
// Kernel 1: fused bf16 WMMA GEMM tile (128 tokens x 128 vocab) + online
// softmax statistics, double-buffered LDS with register prefetch.
// grid = (32 token tiles, 16 vocab groups), block = 256 (8 waves, 2M x 4N).
// ---------------------------------------------------------------------------
__global__ __launch_bounds__(256) void cpo_logit_stats(
    const float* __restrict__ x, const float* __restrict__ W,
    float* __restrict__ pmax, float* __restrict__ psum)
{
  __shared__ __align__(16) unsigned short As[2][TM * AS_STRIDE];
  __shared__ __align__(16) unsigned short Bs[2][TN * AS_STRIDE];
  __shared__ float wmax[4][TM];
  __shared__ float wsum[4][TM];
  __shared__ float rowmax[TM];
  __shared__ float runm[TM];
  __shared__ float runs[TM];

  const int tid  = threadIdx.x;
  const int tok0 = blockIdx.x * TM;
  const int grp  = blockIdx.y;
  const int wave = tid >> 5;
  const int lane = tid & 31;
  const int wm   = wave >> 2;         // 0..1 : 64-row band
  const int wn   = wave & 3;          // 0..3 : 32-col band
  const int half = lane >> 4;
  const int l15  = lane & 15;

  const int t_beg = grp * TILES_PER_GROUP;
  const int t_end = min(t_beg + TILES_PER_GROUP, NTILES);

  if (tid < TM) { runm[tid] = -INFINITY; runs[tid] = 0.0f; }

  for (int nt = t_beg; nt < t_end; ++nt) {
    const int col0 = nt * TN;
    v8f acc[4][2];
    #pragma unroll
    for (int mi = 0; mi < 4; ++mi)
      #pragma unroll
      for (int j = 0; j < 2; ++j)
        #pragma unroll
        for (int e = 0; e < 8; ++e) acc[mi][j][e] = 0.0f;

    float4 pa[4], pb[4];
    // Prologue: fetch + stage K-step 0 into buffer 0.
    global_fetch(x, W, tok0, col0, 0, tid, pa, pb);
    __syncthreads();                    // reuse-guard: prior tile's frag reads done
    lds_store(&As[0][0], &Bs[0][0], tid, pa, pb);

    for (int s = 0; s < NSTEPS; ++s) {
      const int cur = s & 1;
      const int nxt = cur ^ 1;
      // Prefetch next K-step while current buffer is consumed.
      if (s + 1 < NSTEPS)
        global_fetch(x, W, tok0, col0, (s + 1) * KSTEP, tid, pa, pb);

      __syncthreads();                  // buf[cur] staged & buf[nxt] reads drained

      // Preload all fragments, then fire the 8 WMMAs back-to-back.
      const v16bf bf0 = ldsFrag(&Bs[cur][0], wn * 32 + l15,      half);
      const v16bf bf1 = ldsFrag(&Bs[cur][0], wn * 32 + 16 + l15, half);
      v16bf af[4];
      #pragma unroll
      for (int mi = 0; mi < 4; ++mi)
        af[mi] = ldsFrag(&As[cur][0], wm * 64 + mi * 16 + l15, half);
      #pragma unroll
      for (int mi = 0; mi < 4; ++mi) {
        acc[mi][0] = __builtin_amdgcn_wmma_f32_16x16x32_bf16(
            false, af[mi], false, bf0, (short)0, acc[mi][0], false, false);
        acc[mi][1] = __builtin_amdgcn_wmma_f32_16x16x32_bf16(
            false, af[mi], false, bf1, (short)0, acc[mi][1], false, false);
      }

      // Stage next K-step into the other buffer (overlaps with WMMA issue).
      if (s + 1 < NSTEPS)
        lds_store(&As[nxt][0], &Bs[nxt][0], tid, pa, pb);
    }

    // ---- softmax statistics for this 128x128 tile ----
    __syncthreads();
    // (a) per-wave row max over its 32 columns (row striped across 16 lanes)
    #pragma unroll
    for (int mi = 0; mi < 4; ++mi) {
      #pragma unroll
      for (int i = 0; i < 8; ++i) {
        float v = fmaxf(acc[mi][0][i], acc[mi][1][i]);
        #pragma unroll
        for (int off = 8; off >= 1; off >>= 1)
          v = fmaxf(v, __shfl_xor(v, off, 32));
        if (l15 == 0) wmax[wn][wm * 64 + mi * 16 + half * 8 + i] = v;
      }
    }
    __syncthreads();
    // (b) merge the 4 column-band maxes
    if (tid < TM)
      rowmax[tid] = fmaxf(fmaxf(wmax[0][tid], wmax[1][tid]),
                          fmaxf(wmax[2][tid], wmax[3][tid]));
    __syncthreads();
    // (c) per-wave sum-exp against the merged tile max
    #pragma unroll
    for (int mi = 0; mi < 4; ++mi) {
      #pragma unroll
      for (int i = 0; i < 8; ++i) {
        const int row = wm * 64 + mi * 16 + half * 8 + i;
        const float m = rowmax[row];
        float s = __expf(acc[mi][0][i] - m) + __expf(acc[mi][1][i] - m);
        #pragma unroll
        for (int off = 8; off >= 1; off >>= 1)
          s += __shfl_xor(s, off, 32);
        if (l15 == 0) wsum[wn][row] = s;
      }
    }
    __syncthreads();
    // (d) merge tile stats into running (log-sum-exp merge)
    if (tid < TM) {
      const float tm = rowmax[tid];
      const float ts = wsum[0][tid] + wsum[1][tid] + wsum[2][tid] + wsum[3][tid];
      const float rm = runm[tid], rs = runs[tid];
      const float M  = fmaxf(rm, tm);
      runs[tid] = rs * __expf(rm - M) + ts * __expf(tm - M);
      runm[tid] = M;
    }
  }

  __syncthreads();
  if (tid < TM) {
    const size_t idx = (size_t)grp * T_TOTAL + tok0 + tid;
    pmax[idx] = runm[tid];
    psum[idx] = runs[tid];
  }
}

// ---------------------------------------------------------------------------
// Kernel 2: target-token logit, full f32 precision: tgt[t] = x[t] . W[y[t]]
// ---------------------------------------------------------------------------
__global__ __launch_bounds__(256) void cpo_target_dot(
    const float* __restrict__ x, const int* __restrict__ y,
    const float* __restrict__ W, float* __restrict__ tgt)
{
  __shared__ float wred[8];
  const int t   = blockIdx.x;
  const int tid = threadIdx.x;
  const int yv  = y[t];
  const int ys  = (yv == -100) ? 0 : yv;
  const float* xr = x + (size_t)t  * H;
  const float* wr = W + (size_t)ys * H;
  float s = 0.0f;
  #pragma unroll
  for (int k = tid * 4; k < H; k += 256 * 4) {
    const float4 a = *reinterpret_cast<const float4*>(xr + k);
    const float4 b = *reinterpret_cast<const float4*>(wr + k);
    s = fmaf(a.x, b.x, s); s = fmaf(a.y, b.y, s);
    s = fmaf(a.z, b.z, s); s = fmaf(a.w, b.w, s);
  }
  #pragma unroll
  for (int off = 16; off >= 1; off >>= 1) s += __shfl_xor(s, off, 32);
  if ((tid & 31) == 0) wred[tid >> 5] = s;
  __syncthreads();
  if (tid == 0) {
    float r = 0.0f;
    #pragma unroll
    for (int w = 0; w < 8; ++w) r += wred[w];
    tgt[t] = r;
  }
}

// ---------------------------------------------------------------------------
// Kernel 3: merge group partials -> per-token logprob -> CPO scalar loss.
// ---------------------------------------------------------------------------
__global__ __launch_bounds__(256) void cpo_finalize(
    const int* __restrict__ y, const float* __restrict__ pmax,
    const float* __restrict__ psum, const float* __restrict__ tgt,
    float* __restrict__ out)
{
  __shared__ float seq_lp[8];
  __shared__ float seq_cnt[8];
  const int tid = threadIdx.x;
  if (tid < 8) { seq_lp[tid] = 0.0f; seq_cnt[tid] = 0.0f; }
  __syncthreads();

  for (int t = tid; t < T_TOTAL; t += 256) {
    float M = -INFINITY;
    #pragma unroll
    for (int g = 0; g < NGROUPS; ++g)
      M = fmaxf(M, pmax[(size_t)g * T_TOTAL + t]);
    float S = 0.0f;
    #pragma unroll
    for (int g = 0; g < NGROUPS; ++g)
      S += psum[(size_t)g * T_TOTAL + t] * __expf(pmax[(size_t)g * T_TOTAL + t] - M);
    const float lp = tgt[t] - (M + __logf(S));
    if (y[t] != -100) {
      const int b = t >> 9;          // 512 tokens per sequence
      atomicAdd(&seq_lp[b], lp);
      atomicAdd(&seq_cnt[b], 1.0f);
    }
  }
  __syncthreads();

  if (tid == 0) {
    float pref = 0.0f, num = 0.0f, den = 0.0f;
    for (int b = 0; b < 4; ++b) {
      const float c = seq_lp[b]     / seq_cnt[b];
      const float r = seq_lp[b + 4] / seq_cnt[b + 4];
      const float d = 0.1f * (c - r);                       // BETA
      // -log_sigmoid(d) = softplus(-d), numerically stable
      pref += fmaxf(-d, 0.0f) + __logf(1.0f + __expf(-fabsf(d)));
      num  += seq_lp[b];
      den  += seq_cnt[b];
    }
    pref *= 0.25f;                                          // / B
    out[0] = (-num / den) + pref;                           // ALPHA * nll + pref
  }
}

extern "C" void kernel_launch(void* const* d_in, const int* in_sizes, int n_in,
                              void* d_out, int out_size, void* d_ws, size_t ws_size,
                              hipStream_t stream) {
  (void)in_sizes; (void)n_in; (void)out_size; (void)ws_size;
  const float* x = (const float*)d_in[0];
  const int*   y = (const int*)  d_in[1];
  const float* W = (const float*)d_in[2];
  float* out  = (float*)d_out;
  float* pmax = (float*)d_ws;                                // 16*4096 f32
  float* psum = pmax + (size_t)NGROUPS * T_TOTAL;            // 16*4096 f32
  float* tgt  = psum + (size_t)NGROUPS * T_TOTAL;            // 4096 f32

  dim3 g1(T_TOTAL / TM, NGROUPS);
  cpo_logit_stats<<<g1, 256, 0, stream>>>(x, W, pmax, psum);
  cpo_target_dot<<<T_TOTAL, 256, 0, stream>>>(x, y, W, tgt);
  cpo_finalize<<<1, 256, 0, stream>>>(y, pmax, psum, tgt, out);
}
